// PointNet2Multiview2_backup_7954279432401
// MI455X (gfx1250) — compile-verified, matrix operand order checked by reading, not run on hardware
//
#include <hip/hip_runtime.h>
#include <hip/hip_bf16.h>

// ---------------------------------------------------------------------------
// PointNet++ multiview seg for MI455X (gfx1250, wave32).
// MLP matmuls: v_wmma_f32_16x16x32_f16, one wave per 16x16 tile.
//  - Activations staged in f16 with row stride padded to mult-of-32 and
//    zero-filled pads -> branch-free K loop, A fragment = 2x b128 per lane.
//  - Weights pre-packed per layer into exact lane-fragment order -> B
//    fragment = 1x 32B load per lane (coalesced).
//  - BN stats fused into GEMM epilogue via f32 atomics.
// Working set (~120MB) fits the 192MB L2 -> L2-BW/latency bound; irregular
// ops (FPS / ball query / 3NN) stay as VALU kernels.
// ---------------------------------------------------------------------------

constexpr int kB   = 4;
constexpr int kN   = 8192;
constexpr int kV   = 3;
constexpr int kHW  = 32 * 41;   // HF*WF
constexpr int kC   = 128;       // CIMG
constexpr int kCls = 21;
constexpr int kS   = 32;        // nsample

typedef __attribute__((ext_vector_type(16))) _Float16 v16h;
typedef __attribute__((ext_vector_type(8)))  _Float16 v8h;
typedef __attribute__((ext_vector_type(8)))  float    v8f;

static inline int roundup32(int x) { return (x + 31) & ~31; }

// ------------------------------ kernels ------------------------------------

__global__ void transpose_xyz_kernel(const float* __restrict__ xyz,
                                     float* __restrict__ out) {
  int t = blockIdx.x * blockDim.x + threadIdx.x;  // over B*N
  if (t >= kB * kN) return;
  int b = t / kN, n = t % kN;
#pragma unroll
  for (int k = 0; k < 3; ++k)
    out[(long long)t * 3 + k] = xyz[((long long)b * 3 + k) * kN + n];
}

// scatter image features to points, max over views (zero init = reference)
__global__ void project_scatter_kernel(const float* __restrict__ imf,
                                       const int* __restrict__ i3,
                                       const int* __restrict__ i2,
                                       float* __restrict__ out) {
  int t = blockIdx.x * blockDim.x + threadIdx.x;  // over B*V*N
  if (t >= kB * kV * kN) return;
  int j  = t % kN;
  int bv = t / kN;
  int b  = bv / kV;
  const int* p3 = i3 + (long long)bv * (kN + 1);
  const int* p2 = i2 + (long long)bv * (kN + 1);
  int nvalid = p3[0];
  if (j >= nvalid) return;
  int t3 = p3[1 + j], t2 = p2[1 + j];
  const float* src = imf + ((long long)bv * kC) * kHW;
  float* dst = out + ((long long)b * kN + t3) * kC;
  for (int c = 0; c < kC; ++c) {
    float v = src[(long long)c * kHW + t2];
    if (v > 0.f) atomicMax((int*)&dst[c], __float_as_int(v));
  }
}

// farthest point sampling: one block per batch element
__global__ void fps_kernel(const float* __restrict__ xyzsrc, int* __restrict__ fidx,
                           float* __restrict__ dist, int Nsrc, int npoint) {
  __shared__ float sval[256];
  __shared__ int   sidx[256];
  int b = blockIdx.x;
  int t = threadIdx.x;
  const float* p = xyzsrc + (long long)b * Nsrc * 3;
  float* d = dist + (long long)b * Nsrc;
  for (int i = t; i < Nsrc; i += blockDim.x) d[i] = 1e10f;
  __syncthreads();
  int far = 0;
  for (int it = 0; it < npoint; ++it) {
    if (t == 0) fidx[b * npoint + it] = far;
    float cx = p[far * 3], cy = p[far * 3 + 1], cz = p[far * 3 + 2];
    float best = -1.f;
    int bi = 0;
    for (int i = t; i < Nsrc; i += blockDim.x) {
      float dx = p[i * 3] - cx, dy = p[i * 3 + 1] - cy, dz = p[i * 3 + 2] - cz;
      float nd = fminf(d[i], dx * dx + dy * dy + dz * dz);
      d[i] = nd;
      if (nd > best) { best = nd; bi = i; }
    }
    sval[t] = best; sidx[t] = bi;
    __syncthreads();
    for (int s = blockDim.x / 2; s > 0; s >>= 1) {
      if (t < s && sval[t + s] > sval[t]) { sval[t] = sval[t + s]; sidx[t] = sidx[t + s]; }
      __syncthreads();
    }
    far = sidx[0];
    __syncthreads();
  }
}

__global__ void gather3_kernel(const float* __restrict__ xyzsrc,
                               const int* __restrict__ idx,
                               float* __restrict__ out, int Nsrc, int M) {
  int t = blockIdx.x * blockDim.x + threadIdx.x;
  if (t >= kB * M) return;
  int b = t / M;
  int i = idx[t];
  const float* p = xyzsrc + ((long long)b * Nsrc + i) * 3;
  out[t * 3 + 0] = p[0];
  out[t * 3 + 1] = p[1];
  out[t * 3 + 2] = p[2];
}

// ball query + group: one wave32 per center; ordered append via ballot/popc.
// Writes f16 GEMM A rows [rel_xyz | feats | 0-pad] with stride Wpad (mult 32).
__global__ void group_kernel(const float* __restrict__ xyzsrc,
                             const float* __restrict__ newxyz,
                             const float* __restrict__ feats, int C,
                             _Float16* __restrict__ Arows, int Nsrc, int M,
                             int Wpad, float r2) {
  int gt   = blockIdx.x * blockDim.x + threadIdx.x;
  int wid  = gt >> 5;
  int lane = threadIdx.x & 31;
  if (wid >= kB * M) return;
  int b = wid / M;
  const float* px = xyzsrc + (long long)b * Nsrc * 3;
  float cx = newxyz[(long long)wid * 3 + 0];
  float cy = newxyz[(long long)wid * 3 + 1];
  float cz = newxyz[(long long)wid * 3 + 2];
  int Wd = 3 + C;
  long long rowbase = (long long)wid * kS;
  int count = 0;
  for (int basei = 0; basei < Nsrc && count < kS; basei += 32) {
    int i = basei + lane;
    float dx = 0.f, dy = 0.f, dz = 0.f;
    bool in = false;
    if (i < Nsrc) {
      dx = px[i * 3 + 0] - cx;
      dy = px[i * 3 + 1] - cy;
      dz = px[i * 3 + 2] - cz;
      in = (dx * dx + dy * dy + dz * dz) <= r2;
    }
    unsigned mask = (unsigned)__ballot(in);
    int prefix = __popc(mask & ((1u << lane) - 1u));
    int slot = count + prefix;
    if (in && slot < kS) {
      _Float16* row = Arows + (rowbase + slot) * Wpad;
      row[0] = (_Float16)dx; row[1] = (_Float16)dy; row[2] = (_Float16)dz;
      if (feats != nullptr) {
        const float* f = feats + ((long long)b * Nsrc + i) * C;
        for (int c = 0; c < C; ++c) row[3 + c] = (_Float16)f[c];
      }
      for (int c = Wd; c < Wpad; ++c) row[c] = (_Float16)0.f;
    }
    count += __popc(mask);
    if (count > kS) count = kS;
  }
  __threadfence();
  if (count < kS) {  // replicate first hit (center is always in-radius)
    const _Float16* src = Arows + rowbase * Wpad;
    for (int slot = count + lane; slot < kS; slot += 32) {
      _Float16* dstp = Arows + (rowbase + slot) * Wpad;
      for (int c = 0; c < Wpad; ++c) dstp[c] = src[c];
    }
  }
}

// pack f32 weights (Korig x cout) into per-lane WMMA B-fragment order (f16):
// index = ((tc*Ksteps + ks)*32 + lane)*16 + e ; kk = ks*32 + (e<8?e:e+8) + 8*half
__global__ void pack_w_kernel(const float* __restrict__ W, _Float16* __restrict__ Bp,
                              int Korig, int cout, int Ksteps) {
  int t = blockIdx.x * blockDim.x + threadIdx.x;
  int total = ((cout + 15) / 16) * Ksteps * 512;
  if (t >= total) return;
  int e    = t & 15;
  int lane = (t >> 4) & 31;
  int rest = t >> 9;
  int ks   = rest % Ksteps;
  int tc   = rest / Ksteps;
  int half = lane >> 4;
  int n    = tc * 16 + (lane & 15);
  int kk   = ks * 32 + ((e & 8) ? (e + 8) : e) + (half ? 8 : 0);
  float v = (kk < Korig && n < cout) ? W[(long long)kk * cout + n] : 0.f;
  Bp[t] = (_Float16)v;
}

// Core GEMM: one wave32 per 16x16 tile, 8 waves/block. Branch-free K loop:
// A fragment = 2x b128 (16B) loads, B fragment = 1x 32B packed load, then
// v_wmma_f32_16x16x32_f16. Epilogue adds bias, writes f32, optional BN stats.
__global__ void __launch_bounds__(256) wmma_gemm_kernel(
    const _Float16* __restrict__ A, const _Float16* __restrict__ Bp,
    const float* __restrict__ bias, float* __restrict__ out,
    float* __restrict__ sum, float* __restrict__ sumsq,
    int rowTiles, int colTiles, int Ksteps, int Kpad, int cout) {
  int wave = (blockIdx.x * blockDim.x + threadIdx.x) >> 5;
  if (wave >= rowTiles * colTiles) return;
  const int lane = threadIdx.x & 31;
  const int tr = wave % rowTiles;
  const int tc = wave / rowTiles;
  const int half = lane >> 4;
  const int off  = half * 8;
  const int r = tr * 16 + (lane & 15);
  const int n = tc * 16 + (lane & 15);
  const _Float16* arow = A + (long long)r * Kpad + off;
  const _Float16* bptr = Bp + (long long)tc * Ksteps * 512 + lane * 16;
  v8f acc = {};
  for (int ks = 0; ks < Ksteps; ++ks) {
    const _Float16* ap = arow + ks * 32;
    v8h alo = *(const v8h*)(ap);
    v8h ahi = *(const v8h*)(ap + 16);
    v16h b  = *(const v16h*)(bptr + (long long)ks * 512);
    __builtin_prefetch(ap + 32, 0, 1);
    v16h a;
#pragma unroll
    for (int e = 0; e < 8; ++e) { a[e] = alo[e]; a[e + 8] = ahi[e]; }
    acc = __builtin_amdgcn_wmma_f32_16x16x32_f16(false, a, false, b, (short)0,
                                                 acc, false, false);
  }
  const float bb = (n < cout) ? bias[n] : 0.f;
#pragma unroll
  for (int j = 0; j < 8; ++j) {
    const int m = tr * 16 + j + (half ? 8 : 0);  // rows always mult of 16
    if (n < cout) {
      float v = acc[j] + bb;
      out[(long long)m * cout + n] = v;
      if (sum != nullptr) {
        atomicAdd(&sum[n], v);
        atomicAdd(&sumsq[n], v * v);
      }
    }
  }
}

__global__ void bn_finalize_kernel(const float* __restrict__ sum,
                                   const float* __restrict__ sumsq,
                                   const float* __restrict__ gamma,
                                   const float* __restrict__ beta,
                                   float* __restrict__ scale,
                                   float* __restrict__ shift, int C, float invcnt) {
  int c = blockIdx.x * blockDim.x + threadIdx.x;
  if (c >= C) return;
  float mean = sum[c] * invcnt;
  float var  = sumsq[c] * invcnt - mean * mean;
  float sc   = gamma[c] * rsqrtf(var + 1e-5f);
  scale[c] = sc;
  shift[c] = beta[c] - mean * sc;
}

// BN + ReLU, f32 in -> f16 out (feeds next GEMM; C is a multiple of 32)
__global__ void bn_apply_kernel(const float* __restrict__ X,
                                _Float16* __restrict__ Y,
                                const float* __restrict__ scale,
                                const float* __restrict__ shift,
                                long long total, int C) {
  long long t = (long long)blockIdx.x * blockDim.x + threadIdx.x;
  if (t >= total) return;
  int c = (int)(t % C);
  float v = X[t] * scale[c] + shift[c];
  Y[t] = (_Float16)(v > 0.f ? v : 0.f);
}

__global__ void maxpool_kernel(const _Float16* __restrict__ X, float* __restrict__ out,
                               int G, int C, int Wd, int colOff) {
  int t = blockIdx.x * blockDim.x + threadIdx.x;
  if (t >= G * C) return;
  int g = t / C, c = t % C;
  const _Float16* p = X + ((long long)g * kS) * C + c;
  float m = (float)p[0];
  for (int s = 1; s < kS; ++s) m = fmaxf(m, (float)p[(long long)s * C]);
  out[(long long)g * Wd + colOff + c] = m;
}

__global__ void h2f_kernel(const _Float16* __restrict__ src, float* __restrict__ dst,
                           long long n) {
  long long t = (long long)blockIdx.x * blockDim.x + threadIdx.x;
  if (t < n) dst[t] = (float)src[t];
}

// 3-NN inverse-distance interpolation + concat with points1 -> f16 A rows
// (c1+c2 is always a multiple of 32 here, so no pad columns needed)
__global__ void fp_interp_kernel(const float* __restrict__ x1,
                                 const float* __restrict__ x2,
                                 const float* __restrict__ p1,
                                 const float* __restrict__ p2,
                                 _Float16* __restrict__ Arows,
                                 int n1, int s, int c1, int c2) {
  long long t = (long long)blockIdx.x * blockDim.x + threadIdx.x;
  if (t >= (long long)kB * n1) return;
  int b  = (int)(t / n1);
  int Wd = c1 + c2;
  _Float16* row = Arows + t * Wd;
  if (p1 != nullptr)
    for (int c = 0; c < c1; ++c) row[c] = (_Float16)p1[t * c1 + c];
  const float* X2 = x2 + (long long)b * s * 3;
  const float* P2 = p2 + (long long)b * s * (long long)c2;
  float X = x1[t * 3], Y = x1[t * 3 + 1], Z = x1[t * 3 + 2];
  if (s == 1) {
    for (int c = 0; c < c2; ++c) row[c1 + c] = (_Float16)P2[c];
    return;
  }
  float bd0 = 3e38f, bd1 = 3e38f, bd2 = 3e38f;
  int bi0 = 0, bi1 = 0, bi2 = 0;
  for (int j = 0; j < s; ++j) {
    float dx = X - X2[j * 3], dy = Y - X2[j * 3 + 1], dz = Z - X2[j * 3 + 2];
    float d = dx * dx + dy * dy + dz * dz;
    if (d < bd0)      { bd2 = bd1; bi2 = bi1; bd1 = bd0; bi1 = bi0; bd0 = d; bi0 = j; }
    else if (d < bd1) { bd2 = bd1; bi2 = bi1; bd1 = d;   bi1 = j; }
    else if (d < bd2) { bd2 = d;   bi2 = j; }
  }
  float w0 = 1.f / (bd0 + 1e-8f), w1 = 1.f / (bd1 + 1e-8f), w2 = 1.f / (bd2 + 1e-8f);
  float ws = w0 + w1 + w2;
  w0 /= ws; w1 /= ws; w2 /= ws;
  const float* q0 = P2 + (long long)bi0 * c2;
  const float* q1 = P2 + (long long)bi1 * c2;
  const float* q2 = P2 + (long long)bi2 * c2;
  for (int c = 0; c < c2; ++c)
    row[c1 + c] = (_Float16)(w0 * q0[c] + w1 * q1[c] + w2 * q2[c]);
}

// ------------------------------ host side ----------------------------------

struct Layer { const float *W, *b, *beta, *gamma; int cin, cout; };

static void build_layers(void* const* d_in, int base, const int* chans, int nl,
                         Layer* out) {
  for (int i = 0; i < nl; ++i) {
    out[i].W     = (const float*)d_in[base + 4 * i + 0];
    out[i].b     = (const float*)d_in[base + 4 * i + 1];
    out[i].beta  = (const float*)d_in[base + 4 * i + 2];
    out[i].gamma = (const float*)d_in[base + 4 * i + 3];
    out[i].cin = chans[i];
    out[i].cout = chans[i + 1];
  }
}

extern "C" void kernel_launch(void* const* d_in, const int* in_sizes, int n_in,
                              void* d_out, int out_size, void* d_ws, size_t ws_size,
                              hipStream_t stream) {
  (void)in_sizes; (void)n_in; (void)out_size; (void)ws_size;

  const float* xyz = (const float*)d_in[0];
  const float* imf = (const float*)d_in[1];
  const int*   i3  = (const int*)d_in[2];
  const int*   i2  = (const int*)d_in[3];

  // params flattened alphabetically (jax pytree): conv2, fp1..fp4, head,
  // sa1_feat, sa1_geo, sa2_feat, sa2_geo, sa3, sa4; per-layer: W,b,beta,gamma
  const float* conv2W = (const float*)d_in[4];
  const float* conv2b = (const float*)d_in[5];
  static const int ch_fp1[]  = {128, 128, 128, 128};
  static const int ch_fp2[]  = {320, 256, 128};
  static const int ch_fp3[]  = {512, 256, 256};
  static const int ch_fp4[]  = {768, 256, 256};
  static const int ch_head[] = {128, 128};
  static const int ch_s1f[]  = {131, 32, 32, 64};
  static const int ch_s1g[]  = {3, 32, 32, 64};
  static const int ch_s2[]   = {67, 64, 64, 128};
  static const int ch_s3[]   = {259, 128, 128, 256};
  static const int ch_s4[]   = {259, 256, 256, 512};
  Layer fp1L[3];  build_layers(d_in, 6,   ch_fp1, 3, fp1L);
  Layer fp2L[2];  build_layers(d_in, 18,  ch_fp2, 2, fp2L);
  Layer fp3L[2];  build_layers(d_in, 26,  ch_fp3, 2, fp3L);
  Layer fp4L[2];  build_layers(d_in, 34,  ch_fp4, 2, fp4L);
  Layer headL[1]; build_layers(d_in, 42,  ch_head, 1, headL);
  Layer s1fL[3];  build_layers(d_in, 46,  ch_s1f, 3, s1fL);
  Layer s1gL[3];  build_layers(d_in, 58,  ch_s1g, 3, s1gL);
  Layer s2fL[3];  build_layers(d_in, 70,  ch_s2, 3, s2fL);
  Layer s2gL[3];  build_layers(d_in, 82,  ch_s2, 3, s2gL);
  Layer s3L[3];   build_layers(d_in, 94,  ch_s3, 3, s3L);
  Layer s4L[3];   build_layers(d_in, 106, ch_s4, 3, s4L);

  // bump allocator over d_ws
  char* basep = (char*)d_ws;
  size_t off = 0;
  auto alloc = [&](size_t bytes) -> void* {
    void* p = basep + off;
    off += (bytes + 255) & ~(size_t)255;
    return p;
  };
  const size_t F = sizeof(float), H = sizeof(_Float16);
  float* xyzt  = (float*)alloc(F * kB * kN * 3);
  float* imgft = (float*)alloc(F * kB * kN * kC);
  float* dist  = (float*)alloc(F * kB * kN);
  int*   fidx  = (int*)alloc(sizeof(int) * kB * 1024);
  float* l1x   = (float*)alloc(F * kB * 1024 * 3);
  float* l2x   = (float*)alloc(F * kB * 256 * 3);
  float* l3x   = (float*)alloc(F * kB * 64 * 3);
  float* l4x   = (float*)alloc(F * kB * 16 * 3);
  float* l1p   = (float*)alloc(F * kB * 1024 * 64);
  float* l1pf  = (float*)alloc(F * kB * 1024 * 64);
  float* l2p   = (float*)alloc(F * kB * 256 * 256);
  float* l3p   = (float*)alloc(F * kB * 64 * 256);
  float* l4p   = (float*)alloc(F * kB * 16 * 512);
  float* l3p2  = (float*)alloc(F * kB * 64 * 256);
  float* l2p2  = (float*)alloc(F * kB * 256 * 256);
  float* l1p2  = (float*)alloc(F * kB * 1024 * 128);
  _Float16* l0p  = (_Float16*)alloc(H * kB * kN * 128);
  _Float16* bufA = (_Float16*)alloc(H * (size_t)131072 * 160);  // max rows x Kpad
  _Float16* bufB = (_Float16*)alloc(H * (size_t)131072 * 64);   // max rows x cout
  float*    gtmp = (float*)alloc(F * (size_t)131072 * 64);      // pre-BN out
  _Float16* wpk  = (_Float16*)alloc(H * (size_t)262144);        // packed weights
  float* ssum  = (float*)alloc(F * 512);
  float* ssq   = (float*)alloc(F * 512);
  float* scale = (float*)alloc(F * 512);
  float* shift = (float*)alloc(F * 512);

  hipMemsetAsync(imgft, 0, F * kB * kN * kC, stream);
  transpose_xyz_kernel<<<(kB * kN + 255) / 256, 256, 0, stream>>>(xyz, xyzt);
  project_scatter_kernel<<<(kB * kV * kN + 255) / 256, 256, 0, stream>>>(imf, i3, i2, imgft);

  auto launch_gemm = [&](const _Float16* A, const Layer& L, float* outp,
                         float* sm, float* sq, long long rows) {
    int Kpad = roundup32(L.cin);
    int Ksteps = Kpad / 32;
    int colTiles = (L.cout + 15) / 16;
    int packTot = colTiles * Ksteps * 512;
    pack_w_kernel<<<(packTot + 255) / 256, 256, 0, stream>>>(L.W, wpk, L.cin,
                                                             L.cout, Ksteps);
    int rowTiles = (int)(rows / 16);
    long long waves = (long long)rowTiles * colTiles;
    wmma_gemm_kernel<<<(unsigned)((waves * 32 + 255) / 256), 256, 0, stream>>>(
        A, wpk, L.b, outp, sm, sq, rowTiles, colTiles, Ksteps, Kpad, L.cout);
  };

  // MLP stack: pack-W -> GEMM(+bias,+stats) -> BN finalize -> BN+ReLU (f16)
  auto run_mlp = [&](const Layer* Ls, int nl, const _Float16* input,
                     long long rows) -> const _Float16* {
    const _Float16* cur = input;
    _Float16* t0 = (input == bufB) ? bufA : bufB;
    _Float16* t1 = (t0 == bufB) ? bufA : bufB;
    for (int i = 0; i < nl; ++i) {
      hipMemsetAsync(ssum, 0, F * 512, stream);
      hipMemsetAsync(ssq, 0, F * 512, stream);
      launch_gemm(cur, Ls[i], gtmp, ssum, ssq, rows);
      bn_finalize_kernel<<<1, 512, 0, stream>>>(ssum, ssq, Ls[i].gamma, Ls[i].beta,
                                                scale, shift, Ls[i].cout,
                                                1.0f / (float)rows);
      _Float16* dst = (i & 1) ? t1 : t0;
      long long tot = rows * Ls[i].cout;
      bn_apply_kernel<<<(unsigned)((tot + 255) / 256), 256, 0, stream>>>(
          gtmp, dst, scale, shift, tot, Ls[i].cout);
      cur = dst;
    }
    return cur;
  };

  // set-abstraction: FPS -> gather -> ball-group(f16,padded) -> MLP -> maxpool
  auto run_sa = [&](const float* xs, int Nsrc, const float* feats, int C, int M,
                    float radius, const Layer* Ls, int nl, float* outxyz,
                    float* outp, int outW, int colOff) {
    fps_kernel<<<kB, 256, 0, stream>>>(xs, fidx, dist, Nsrc, M);
    gather3_kernel<<<(kB * M + 255) / 256, 256, 0, stream>>>(xs, fidx, outxyz, Nsrc, M);
    int Wpad = roundup32(3 + C);
    group_kernel<<<(kB * M * 32 + 255) / 256, 256, 0, stream>>>(
        xs, outxyz, feats, C, bufA, Nsrc, M, Wpad, radius * radius);
    long long rows = (long long)kB * M * kS;
    const _Float16* fin = run_mlp(Ls, nl, bufA, rows);
    int G = kB * M, Cc = Ls[nl - 1].cout;
    maxpool_kernel<<<(G * Cc + 255) / 256, 256, 0, stream>>>(fin, outp, G, Cc, outW, colOff);
  };

  // feature propagation: 3NN interp + concat -> MLP -> persistent copy
  auto run_fp = [&](const float* x1, int n1, const float* x2, int s,
                    const float* p1, int c1, const float* p2, int c2,
                    const Layer* Ls, int nl, float* outp_f32,
                    _Float16* outp_f16) {
    long long rows = (long long)kB * n1;
    fp_interp_kernel<<<(unsigned)((rows + 255) / 256), 256, 0, stream>>>(
        x1, x2, p1, p2, bufA, n1, s, c1, c2);
    const _Float16* fin = run_mlp(Ls, nl, bufA, rows);
    long long tot = rows * Ls[nl - 1].cout;
    if (outp_f32 != nullptr)
      h2f_kernel<<<(unsigned)((tot + 255) / 256), 256, 0, stream>>>(fin, outp_f32, tot);
    else
      hipMemcpyAsync(outp_f16, (const void*)fin, H * tot,
                     hipMemcpyDeviceToDevice, stream);
  };

  run_sa(xyzt, kN,   nullptr, 0,   1024, 0.1f, s1gL, 3, l1x, l1p,  64,  0);
  run_sa(xyzt, kN,   imgft,   128, 1024, 0.1f, s1fL, 3, l1x, l1pf, 64,  0);
  run_sa(l1x,  1024, l1p,     64,  256,  0.2f, s2gL, 3, l2x, l2p,  256, 0);
  run_sa(l1x,  1024, l1pf,    64,  256,  0.2f, s2fL, 3, l2x, l2p,  256, 128);
  run_sa(l2x,  256,  l2p,     256, 64,   0.4f, s3L,  3, l3x, l3p,  256, 0);
  run_sa(l3x,  64,   l3p,     256, 16,   0.8f, s4L,  3, l4x, l4p,  512, 0);

  run_fp(l3x,  64,   l4x, 16,   l3p,     256, l4p,  512, fp4L, 2, l3p2, nullptr);
  run_fp(l2x,  256,  l3x, 64,   l2p,     256, l3p2, 256, fp3L, 2, l2p2, nullptr);
  run_fp(l1x,  1024, l2x, 256,  l1p,     64,  l2p2, 256, fp2L, 2, l1p2, nullptr);
  run_fp(xyzt, kN,   l1x, 1024, nullptr, 0,   l1p2, 128, fp1L, 3, nullptr, l0p);

  const _Float16* hout = run_mlp(headL, 1, l0p, (long long)kB * kN);

  // final classifier (bias only, no BN/ReLU) straight into d_out (f32)
  Layer convL;
  convL.W = conv2W; convL.b = conv2b; convL.beta = nullptr; convL.gamma = nullptr;
  convL.cin = 128; convL.cout = kCls;
  launch_gemm(hout, convL, (float*)d_out, nullptr, nullptr, (long long)kB * kN);
}
